// CrossAttentionFusion_73212012527626
// MI455X (gfx1250) — matrix operand order repaired and checked by reading, not compiled, not run on hardware
//
#include <hip/hip_runtime.h>
#include <hip/hip_bf16.h>

typedef __attribute__((ext_vector_type(16))) _Float16 v16h;
typedef __attribute__((ext_vector_type(8)))  _Float16 v8h;
typedef __attribute__((ext_vector_type(4)))  _Float16 v4h;
typedef __attribute__((ext_vector_type(8)))  float    v8f;

#define NQ   4096   // tokens per batch (16*16*16)
#define DIMQ 256
#define DIMK 32
#define KT   32     // k-tokens per attention iteration
#define MQ   64     // queries per attention workgroup (4 waves)
#define NSPLIT 2    // split-K over key tokens
#define KSPAN (NQ / NSPLIT)

// ---- async copy: global -> LDS (16B per lane), tracked by ASYNCcnt --------
__device__ __forceinline__ void async_copy_b128(_Float16* lds_dst, const _Float16* gsrc) {
  unsigned int       l = (unsigned int)(unsigned long long)lds_dst;  // low 32b = LDS offset
  unsigned long long g = (unsigned long long)gsrc;
  asm volatile("global_load_async_to_lds_b128 %0, %1, off"
               :: "v"(l), "v"(g) : "memory");
}
__device__ __forceinline__ void wait_async_le16() {
  asm volatile("s_wait_asynccnt 0x10" ::: "memory");
}
__device__ __forceinline__ void wait_async_0() {
  asm volatile("s_wait_asynccnt 0x0" ::: "memory");
}

// ---- WMMA fragment helpers (gfx1250 wave32 layouts per CDNA5 ISA 7.12.2) ----

// A fragment 16x32 f16 from row-major f16 [row][ld]:
// lane<16: M=lane, K halves {kc..kc+7, kc+16..kc+23}
// lane>=16: M=lane-16, K halves {kc+8..kc+15, kc+24..kc+31}
__device__ __forceinline__ v16h load_A16(const _Float16* base, int ld, int kc, int lane) {
  int row = lane & 15;
  int kb  = kc + ((lane >> 4) << 3);
  const _Float16* p = base + row * ld + kb;
  v8h lo = *(const v8h*)(p);
  v8h hi = *(const v8h*)(p + 16);
  v16h a;
#pragma unroll
  for (int i = 0; i < 8; ++i) { a[i] = lo[i]; a[i + 8] = hi[i]; }
  return a;
}

// B fragment 32x16 f16 where memory is [n(col)][K contiguous] row-major, ld halves:
// lane<16: N=lane, K {kc..kc+7, kc+16..kc+23} ; lane>=16: N=lane-16, K {kc+8..kc+15, kc+24..kc+31}
__device__ __forceinline__ v16h load_B16(const _Float16* base, int ld, int kc, int lane) {
  int col = lane & 15;
  int kb  = kc + ((lane >> 4) << 3);
  const _Float16* p = base + col * ld + kb;
  v8h lo = *(const v8h*)(p);
  v8h hi = *(const v8h*)(p + 16);
  v16h b;
#pragma unroll
  for (int i = 0; i < 8; ++i) { b[i] = lo[i]; b[i + 8] = hi[i]; }
  return b;
}

__device__ __forceinline__ v8f wmma_f16(v16h a, v16h b, v8f c) {
  return __builtin_amdgcn_wmma_f32_16x16x32_f16(false, a, false, b, (short)0, c,
                                                false, false);
}

// ---------------------------------------------------------------------------
// Kernel 0: one-shot f32 -> f16 conversion of all four weight matrices.
// Regions (float4 units): Wq 16384, Wk 2048, Wv 2048, Wo 16384 (total 36864)
// ---------------------------------------------------------------------------
__global__ __launch_bounds__(256) void wcvt_kernel(
    const float* __restrict__ s0, const float* __restrict__ s1,
    const float* __restrict__ s2, const float* __restrict__ s3,
    _Float16* __restrict__ d0, _Float16* __restrict__ d1,
    _Float16* __restrict__ d2, _Float16* __restrict__ d3)
{
  int i = blockIdx.x * 256 + threadIdx.x;
  const float* s; _Float16* d; int off;
  if      (i < 16384) { s = s0; d = d0; off = i; }
  else if (i < 18432) { s = s1; d = d1; off = i - 16384; }
  else if (i < 20480) { s = s2; d = d2; off = i - 18432; }
  else                { s = s3; d = d3; off = i - 20480; }
  float4 v = ((const float4*)s)[off];
  v4h o;
  o[0] = (_Float16)v.x; o[1] = (_Float16)v.y;
  o[2] = (_Float16)v.z; o[3] = (_Float16)v.w;
  ((v4h*)d)[off] = o;
}

// ---------------------------------------------------------------------------
// Kernel 1: projection  out[o, n-tile] = W[o, :] . X[:, n]  (+bias)*scale
// X is channel-major f32 [cdim][NQ] per batch; W is pre-converted f16.
//   transposed==0 : token-major  out[(b*NQ+n)*256 + o]
//   transposed==1 : dim-major    out[(b*256+o)*NQ + n]   (used for V^T)
// ---------------------------------------------------------------------------
__global__ __launch_bounds__(256) void proj_kernel(
    const float* __restrict__ X, const _Float16* __restrict__ Wh,
    const float* __restrict__ bias, _Float16* __restrict__ out,
    int cdim, float scale, int transposed)
{
  __shared__ __align__(32) _Float16 X_lds[16 * DIMQ];  // [token][c], <= 8KB
  const int b   = blockIdx.y;
  const int n0  = blockIdx.x * 16;
  const int tid = threadIdx.x;
  const float* Xb = X + (size_t)b * cdim * NQ;

  // stage + transpose X tile: LDS[token][c] = X[c][n0+token]  (f32 -> f16)
  for (int c = tid; c < cdim; c += 256) {
    const float* src = Xb + (size_t)c * NQ + n0;
#pragma unroll
    for (int nt = 0; nt < 16; ++nt)
      X_lds[nt * cdim + c] = (_Float16)src[nt];
  }
  __syncthreads();

  const int wave = tid >> 5;
  const int lane = tid & 31;
  const int hi   = lane >> 4;

#pragma unroll
  for (int sub = 0; sub < 2; ++sub) {
    const int o0 = (wave * 2 + sub) * 16;
    v8f acc = {};
    for (int kc = 0; kc < cdim; kc += 32) {
      v16h a  = load_A16(Wh + (size_t)o0 * cdim, cdim, kc, lane);
      v16h bf = load_B16(X_lds, cdim, kc, lane);
      acc = wmma_f16(a, bf, acc);
    }
    // D fragment: M = o0 + hi*8 + r, N = token = n0 + (lane&15)
    const int tok  = n0 + (lane & 15);
    const int orow = o0 + hi * 8;
    if (!transposed) {
      v8h st;
#pragma unroll
      for (int r = 0; r < 8; ++r)
        st[r] = (_Float16)((acc[r] + bias[orow + r]) * scale);
      *(v8h*)(out + ((size_t)b * NQ + tok) * DIMQ + orow) = st;   // 16B contig
    } else {
#pragma unroll
      for (int r = 0; r < 8; ++r)
        out[((size_t)b * DIMQ + orow + r) * NQ + tok] =
            (_Float16)((acc[r] + bias[orow + r]) * scale);
    }
  }
}

// ---------------------------------------------------------------------------
// Kernel 2: flash attention, split-K. 4 waves / 64 queries per WG; each WG
// streams KSPAN=2048 k-tokens (KT=32 per iter), double-buffered via
// global_load_async_to_lds_b128 (ASYNCcnt). Emits UN-normalized partial O
// (f32) + per-row (m, l) stats; merge kernel combines the splits.
// ---------------------------------------------------------------------------
__device__ __forceinline__ void stage_kv_async(
    _Float16* Kb, _Float16* Vb,
    const _Float16* Kg, const _Float16* Vg, int k0, int tid)
{
  const _Float16* ks = Kg + (size_t)k0 * DIMQ;
  // K tile: 32 token rows x 256 d, contiguous 16KB   (8 x b128 per thread)
#pragma unroll
  for (int i = tid; i < KT * DIMQ / 8; i += 128)
    async_copy_b128(Kb + i * 8, ks + i * 8);
  // V tile: V^T rows (d-major), 64B per d row        (8 x b128 per thread)
#pragma unroll
  for (int i = tid; i < DIMQ * KT / 8; i += 128) {
    int d = i >> 2, c = i & 3;
    async_copy_b128(Vb + i * 8, Vg + (size_t)d * NQ + k0 + c * 8);
  }
}

__global__ __launch_bounds__(128) void attn_kernel(
    const _Float16* __restrict__ Qh, const _Float16* __restrict__ Kh,
    const _Float16* __restrict__ Vt, float* __restrict__ Opart,
    float* __restrict__ Ms, float* __restrict__ Ls)
{
  __shared__ __align__(32) _Float16 Q_lds[MQ * DIMQ];      // 32 KB [q][d]
  __shared__ __align__(32) _Float16 K_lds[2][KT * DIMQ];   // 2 x 16 KB [kt][d]
  __shared__ __align__(32) _Float16 V_lds[2][DIMQ * KT];   // 2 x 16 KB [d][kt]
  __shared__ __align__(32) _Float16 P_lds[4][16 * KT];     // 4 KB per-wave [q][kt]

  const int q0    = blockIdx.x * MQ;
  const int split = blockIdx.y;
  const int b     = blockIdx.z;
  const int tid   = threadIdx.x;
  const int wave  = tid >> 5;
  const int lane  = tid & 31;
  const int kbase = split * KSPAN;
  const int sb    = split * 2 + b;   // partial-buffer slot

  const _Float16* Kg = Kh + (size_t)b * NQ * DIMQ;
  const _Float16* Vg = Vt + (size_t)b * DIMQ * NQ;

  // async stage Q strip (contiguous 32 KB) + first K/V tile
  {
    const _Float16* qs = Qh + ((size_t)b * NQ + q0) * DIMQ;
#pragma unroll
    for (int i = tid; i < MQ * DIMQ / 8; i += 128)
      async_copy_b128(Q_lds + i * 8, qs + i * 8);
  }
  stage_kv_async(K_lds[0], V_lds[0], Kg, Vg, kbase, tid);

  v8f o_acc[16];
  v8f vzero = {};
#pragma unroll
  for (int t = 0; t < 16; ++t) o_acc[t] = vzero;
  float m_i[8], l_i[8];
#pragma unroll
  for (int r = 0; r < 8; ++r) { m_i[r] = -3.0e38f; l_i[r] = 0.0f; }

  const _Float16* Qw = Q_lds + wave * 16 * DIMQ;
  _Float16*       Pw = P_lds[wave];

  const int NITER = KSPAN / KT;   // 64
  for (int it = 0; it < NITER; ++it) {
    const int buf = it & 1;
    // issue next tile into the other buffer, then wait for current tile:
    // async ops complete in order; each tile = 16 ops/thread, so <=16
    // outstanding implies everything before the last tile has landed.
    if (it + 1 < NITER) {
      stage_kv_async(K_lds[buf ^ 1], V_lds[buf ^ 1], Kg, Vg,
                     kbase + (it + 1) * KT, tid);
      wait_async_le16();
    } else {
      wait_async_0();
    }
    __syncthreads();

    const _Float16* Kb = K_lds[buf];
    const _Float16* Vb = V_lds[buf];

    // ---- S = Q . K^T : two 16x16 tiles over ktok, contraction over d=256 ----
    v8f S0 = vzero, S1 = vzero;
#pragma unroll
    for (int kc = 0; kc < DIMQ; kc += 32) {
      v16h a  = load_A16(Qw, DIMQ, kc, lane);
      v16h b0 = load_B16(Kb,             DIMQ, kc, lane);  // ktok 0..15
      v16h b1 = load_B16(Kb + 16 * DIMQ, DIMQ, kc, lane);  // ktok 16..31
      S0 = wmma_f16(a, b0, S0);
      S1 = wmma_f16(a, b1, S1);
    }

    // ---- online softmax (base 2). Row r lives in a 16-lane half-group. ----
#pragma unroll
    for (int r = 0; r < 8; ++r) {
      float mx = fmaxf(S0[r], S1[r]);
#pragma unroll
      for (int mask = 1; mask <= 8; mask <<= 1)
        mx = fmaxf(mx, __shfl_xor(mx, mask, 32));
      float mn    = fmaxf(m_i[r], mx);
      float alpha = exp2f(m_i[r] - mn);
      float p0    = exp2f(S0[r] - mn);
      float p1    = exp2f(S1[r] - mn);
      float rs    = p0 + p1;
#pragma unroll
      for (int mask = 1; mask <= 8; mask <<= 1)
        rs += __shfl_xor(rs, mask, 32);
      l_i[r] = l_i[r] * alpha + rs;
      m_i[r] = mn;
      S0[r] = p0;
      S1[r] = p1;
#pragma unroll
      for (int t = 0; t < 16; ++t) o_acc[t][r] *= alpha;
    }

    // ---- P: D-fragment -> LDS (per-wave, in-order LDS, no barrier) ----
    {
      const int colA   = lane & 15;
      const int rowoff = (lane >> 4) * 8;
#pragma unroll
      for (int r = 0; r < 8; ++r) {
        Pw[(rowoff + r) * KT + colA]      = (_Float16)S0[r];
        Pw[(rowoff + r) * KT + 16 + colA] = (_Float16)S1[r];
      }
    }
    v16h pa = load_A16(Pw, KT, 0, lane);
    // ---- O += P . V over 16 d-tiles ----
#pragma unroll
    for (int t = 0; t < 16; ++t) {
      v16h vb = load_B16(Vb + t * 16 * KT, KT, 0, lane);
      o_acc[t] = wmma_f16(pa, vb, o_acc[t]);
    }
    __syncthreads();   // all waves done with buf before it is refilled
  }

  // epilogue: store UN-normalized partial O (f32) + row stats
  const int colD   = lane & 15;
  const int rowoff = (lane >> 4) * 8;
  float* Ob = Opart + ((size_t)sb * NQ + q0 + wave * 16) * DIMQ;
#pragma unroll
  for (int t = 0; t < 16; ++t) {
#pragma unroll
    for (int r = 0; r < 8; ++r)
      Ob[(size_t)(rowoff + r) * DIMQ + t * 16 + colD] = o_acc[t][r];
  }
  if ((lane & 15) == 0) {
    const size_t srow = (size_t)sb * NQ + q0 + wave * 16 + rowoff;
#pragma unroll
    for (int r = 0; r < 8; ++r) {
      Ms[srow + r] = m_i[r];
      Ls[srow + r] = l_i[r];
    }
  }
}

// ---------------------------------------------------------------------------
// Kernel 2b: merge the NSPLIT partial results -> normalized f16 Ah [b][q][d]
// ---------------------------------------------------------------------------
__global__ __launch_bounds__(256) void attn_merge_kernel(
    const float* __restrict__ Opart, const float* __restrict__ Ms,
    const float* __restrict__ Ls, _Float16* __restrict__ Ah)
{
  // one thread per float4 group: B*NQ*DIMQ/4 = 524288
  int gid = blockIdx.x * 256 + threadIdx.x;
  int d4  = gid & 63;
  int q   = (gid >> 6) & (NQ - 1);
  int b   = gid >> 18;

  size_t r0 = (size_t)(0 * 2 + b) * NQ + q;
  size_t r1 = (size_t)(1 * 2 + b) * NQ + q;
  float m0 = Ms[r0], m1 = Ms[r1];
  float l0 = Ls[r0], l1 = Ls[r1];
  float m  = fmaxf(m0, m1);
  float w0 = exp2f(m0 - m);
  float w1 = exp2f(m1 - m);
  float inv = 1.0f / (w0 * l0 + w1 * l1);
  w0 *= inv; w1 *= inv;

  float4 o0 = ((const float4*)(Opart + r0 * DIMQ))[d4];
  float4 o1 = ((const float4*)(Opart + r1 * DIMQ))[d4];
  v4h o;
  o[0] = (_Float16)(w0 * o0.x + w1 * o1.x);
  o[1] = (_Float16)(w0 * o0.y + w1 * o1.y);
  o[2] = (_Float16)(w0 * o0.z + w1 * o1.z);
  o[3] = (_Float16)(w0 * o0.w + w1 * o1.w);
  ((v4h*)(Ah + ((size_t)b * NQ + q) * DIMQ))[d4] = o;
}

// ---------------------------------------------------------------------------
// Kernel 3: out = F_VNet + (Ah . Wo^T + bo), f32 output channel-major [b][c][n]
// ---------------------------------------------------------------------------
__global__ __launch_bounds__(256) void out_proj_kernel(
    const _Float16* __restrict__ Ah, const _Float16* __restrict__ Woh,
    const float* __restrict__ bo, const float* __restrict__ Fv,
    float* __restrict__ out)
{
  __shared__ __align__(32) _Float16 A_lds[16 * DIMQ];  // 8 KB [token][c]
  const int b   = blockIdx.y;
  const int n0  = blockIdx.x * 16;
  const int tid = threadIdx.x;
  {
    const _Float16* src = Ah + ((size_t)b * NQ + n0) * DIMQ;
#pragma unroll
    for (int i = tid; i < 16 * DIMQ / 8; i += 256)
      async_copy_b128(A_lds + i * 8, src + i * 8);
  }
  wait_async_0();
  __syncthreads();
  const int wave = tid >> 5, lane = tid & 31, hi = lane >> 4;
#pragma unroll
  for (int sub = 0; sub < 2; ++sub) {
    const int o0 = (wave * 2 + sub) * 16;
    v8f acc = {};
#pragma unroll
    for (int kc = 0; kc < DIMQ; kc += 32) {
      v16h a  = load_A16(Woh + (size_t)o0 * DIMQ, DIMQ, kc, lane);
      v16h bf = load_B16(A_lds, DIMQ, kc, lane);
      acc = wmma_f16(a, bf, acc);
    }
    const int tok  = n0 + (lane & 15);
    const int orow = o0 + hi * 8;
#pragma unroll
    for (int r = 0; r < 8; ++r) {
      size_t idx = ((size_t)b * DIMQ + orow + r) * NQ + tok;
      out[idx] = Fv[idx] + acc[r] + bo[orow + r];
    }
  }
}

// ---------------------------------------------------------------------------
extern "C" void kernel_launch(void* const* d_in, const int* in_sizes, int n_in,
                              void* d_out, int out_size, void* d_ws, size_t ws_size,
                              hipStream_t stream) {
  (void)in_sizes; (void)n_in; (void)out_size; (void)ws_size;
  const float* Fv = (const float*)d_in[0];
  const float* Fk = (const float*)d_in[1];
  const float* Wq = (const float*)d_in[2];
  const float* bq = (const float*)d_in[3];
  const float* Wk = (const float*)d_in[4];
  const float* bk = (const float*)d_in[5];
  const float* Wv = (const float*)d_in[6];
  const float* bv = (const float*)d_in[7];
  const float* Wo = (const float*)d_in[8];
  const float* bo = (const float*)d_in[9];
  float* out = (float*)d_out;

  // workspace: 16 MiB f16 activations + 288 KB f16 weights
  //          + 16 MiB f32 split-K partials + 128 KB stats   (~32.4 MiB)
  const size_t ACT = (size_t)2 * NQ * DIMQ;
  _Float16* Qh  = (_Float16*)d_ws;
  _Float16* Kh  = Qh + ACT;
  _Float16* Vt  = Kh + ACT;
  _Float16* Ah  = Vt + ACT;
  _Float16* Wqh = Ah + ACT;
  _Float16* Wkh = Wqh + (size_t)DIMQ * DIMQ;
  _Float16* Wvh = Wkh + (size_t)DIMQ * DIMK;
  _Float16* Woh = Wvh + (size_t)DIMQ * DIMK;
  float*    Opart = (float*)(Woh + (size_t)DIMQ * DIMQ);  // [NSPLIT*2][NQ][DIMQ]
  float*    Ms    = Opart + (size_t)NSPLIT * ACT;
  float*    Ls    = Ms + (size_t)NSPLIT * 2 * NQ;

  const float LOG2E   = 1.44269504088896340736f;
  const float scale_q = 0.125f * LOG2E;  // (256/4)^-0.5 folded with log2(e)

  wcvt_kernel<<<144, 256, 0, stream>>>(Wq, Wk, Wv, Wo, Wqh, Wkh, Wvh, Woh);

  dim3 gproj(NQ / 16, 2);
  proj_kernel<<<gproj, 256, 0, stream>>>(Fv, Wqh, bq, Qh, DIMQ, scale_q, 0);
  proj_kernel<<<gproj, 256, 0, stream>>>(Fk, Wkh, bk, Kh, DIMK, 1.0f, 0);
  proj_kernel<<<gproj, 256, 0, stream>>>(Fk, Wvh, bv, Vt, DIMK, 1.0f, 1);

  dim3 gattn(NQ / MQ, NSPLIT, 2);
  attn_kernel<<<gattn, 128, 0, stream>>>(Qh, Kh, Vt, Opart, Ms, Ls);

  attn_merge_kernel<<<(2 * NQ * DIMQ / 4) / 256, 256, 0, stream>>>(Opart, Ms, Ls, Ah);

  out_proj_kernel<<<gproj, 256, 0, stream>>>(Ah, Woh, bo, Fv, out);
}
